// GCN_2362232013007
// MI455X (gfx1250) — matrix operand order repaired and checked by reading, not compiled, not run on hardware
//
#include <hip/hip_runtime.h>
#include <hip/hip_bf16.h>

// CDNA5 / gfx1250 GNN implementation, round 4.
// - 256-thread block = 8 wave32 (CDNA5!): tile group = 128 rows, 8 act tiles.
// - All MLP layers on v_wmma_f32_16x16x32_f16 (f16 in, f32 accumulate).
// - Weights pre-swizzled in LDS to B-fragment order: 2x ds_load_b128 per frag,
//   software-pipelined one WMMA ahead. Biases in LDS. lrelu = max(x, 0.01x).
// - Edge kernel: dst-gather held in registers across the k1 MLP to overlap
//   global latency; edge-feat rows prefetched (global_prefetch_b8).

typedef __attribute__((ext_vector_type(16))) _Float16 v16h;
typedef __attribute__((ext_vector_type(8)))  _Float16 v8h;
typedef __attribute__((ext_vector_type(8)))  float    v8f;

#define GN_NODES 50000
#define GN_EDGES 800000
#define GN_NFEAT 32
#define GN_EFEAT 16
#define GN_NOUT  32
#define GN_EOUT  16

constexpr int ACT_LD    = 72;            // halves; 144 B row stride (16B aligned, bank-friendly)
constexpr int ACT_TILE  = 16 * ACT_LD;   // 1152 halves per wave tile
constexpr int WAVES_PB  = 8;             // 256 threads / wave32
constexpr int ROWS_PB   = WAVES_PB * 16; // 128 rows per block group

struct MlpW  { const float *W0,*b0,*W1,*b1,*W2,*b2; };                  // global f32
struct MlpWS { const _Float16 *W0,*W1,*W2; const float *b0,*b1,*b2; };  // LDS f16 + LDS f32 bias

// leaky_relu(x) = max(x, 0.01*x): v_mul + v_max, dual-issuable, no VCC.
__device__ __forceinline__ float lrelu(float v) { return fmaxf(v, 0.01f * v); }

__device__ __forceinline__ void lds_fence() {
  asm volatile("s_wait_dscnt 0" ::: "memory");
}

// ---- fragment builders -----------------------------------------------------
// A fragment: 16x32 f16 tile from wave-private LDS act[16][ACT_LD].
__device__ __forceinline__ v16h load_a_frag(const _Float16* act, int lane, int kbase) {
  const int row = lane & 15;
  const int hi  = lane >> 4;
  const _Float16* base = act + row * ACT_LD;
  v8h lo  = *(const v8h*)(base + kbase + hi * 8);
  v8h hi8 = *(const v8h*)(base + kbase + 16 + hi * 8);
  return __builtin_shufflevector(lo, hi8, 0,1,2,3,4,5,6,7,8,9,10,11,12,13,14,15);
}

// B fragment from swizzled LDS weights: fragment = 512 contiguous halves,
// lane-major (32 lanes x 16 halves). Two ds_load_b128 per lane.
__device__ __forceinline__ v16h load_b_frag(const _Float16* W, int lane, int frag) {
  const _Float16* base = W + frag * 512 + lane * 16;
  v8h lo  = *(const v8h*)(base);
  v8h hi8 = *(const v8h*)(base + 8);
  return __builtin_shufflevector(lo, hi8, 0,1,2,3,4,5,6,7,8,9,10,11,12,13,14,15);
}

// ---- one GEMM layer: D[16xDOUT] = act[16xDINP] * W + bias ------------------
template<int DINP, int DOUT>
__device__ __forceinline__ void layer_frags(const _Float16* act, const _Float16* W,
                                            const float* bias, int lane, v8f* fr) {
  constexpr int KT = DINP / 32;
  constexpr int NT = DOUT / 16;
  v16h a[KT];
#pragma unroll
  for (int kt = 0; kt < KT; ++kt) a[kt] = load_a_frag(act, lane, kt * 32);
  // stored fragment index for flat iteration i (nt-major): frag = (i%KT)*NT + i/KT
  v16h bcur = load_b_frag(W, lane, 0);
#pragma unroll
  for (int nt = 0; nt < NT; ++nt) {
    v8f c = {};
#pragma unroll
    for (int kt = 0; kt < KT; ++kt) {
      v16h bnext = bcur;
      const int i = nt * KT + kt + 1;
      if (i < NT * KT) bnext = load_b_frag(W, lane, (i % KT) * NT + (i / KT));
      c = __builtin_amdgcn_wmma_f32_16x16x32_f16(false, a[kt], false, bcur,
                                                 (short)0, c, false, false);
      bcur = bnext;
    }
    const float bv = bias[nt * 16 + (lane & 15)];   // LDS f32
#pragma unroll
    for (int r = 0; r < 8; ++r) c[r] += bv;
    fr[nt] = c;
  }
}

// C/D layout -> LDS act tile (optionally leaky_relu), f16.
__device__ __forceinline__ void frags_to_act(_Float16* act, int lane,
                                             const v8f* fr, int nfr, bool relu) {
  const int col0 = lane & 15;
  const int row0 = (lane >> 4) << 3;
  for (int nt = 0; nt < nfr; ++nt) {
#pragma unroll
    for (int r = 0; r < 8; ++r) {
      float v = fr[nt][r];
      if (relu) v = lrelu(v);
      act[(row0 + r) * ACT_LD + nt * 16 + col0] = (_Float16)v;
    }
  }
  lds_fence();
}

__device__ __forceinline__ void frags_to_global(float* out, long rowbase, long nrows,
                                                int ldo, int lane, const v8f* fr, int nfr) {
  const int col0 = lane & 15;
  const int row0 = (lane >> 4) << 3;
  for (int nt = 0; nt < nfr; ++nt) {
#pragma unroll
    for (int r = 0; r < 8; ++r) {
      long row = rowbase + row0 + r;
      if (row < nrows) out[row * ldo + nt * 16 + col0] = fr[nt][r];
    }
  }
}

// ---- staging ---------------------------------------------------------------
// Global f32 weights [din][dout] -> LDS f16 in B-fragment-native order.
__device__ __forceinline__ void stage_weights(const float* g, _Float16* s,
                                              int din, int dinp, int dout) {
  const int NT = dout >> 4;
  for (int i = threadIdx.x; i < dinp * dout; i += blockDim.x) {
    const int k = i / dout, n = i - k * dout;
    const int kt = k >> 5, nt = n >> 4;
    const int lane = (n & 15) + (((k >> 3) & 1) << 4);
    const int j = (k & 7) + (((k >> 4) & 1) << 3);
    const int pos = (kt * NT + nt) * 512 + lane * 16 + j;
    s[pos] = (k < din) ? (_Float16)g[i] : (_Float16)0.0f;
  }
}

__device__ __forceinline__ void stage_bias(const float* g, float* s, int n) {
  for (int i = threadIdx.x; i < n; i += blockDim.x) s[i] = g[i];
}

__device__ __forceinline__ v8h cvt8v(float4 f0, float4 f1) {
  v8h h;
  h[0] = (_Float16)f0.x; h[1] = (_Float16)f0.y; h[2] = (_Float16)f0.z; h[3] = (_Float16)f0.w;
  h[4] = (_Float16)f1.x; h[5] = (_Float16)f1.y; h[6] = (_Float16)f1.z; h[7] = (_Float16)f1.w;
  return h;
}

// 16 contiguous rows of [nrows][nf] f32 -> act[16][ACT_LD] f16, zero-pad cols >= nf.
__device__ __forceinline__ void stage_rows(const float* src, long rowbase, long nrows,
                                           int nf, _Float16* act, int lane) {
  const int row  = lane >> 1;
  const int half = lane & 1;
  long r = rowbase + row;
  if (r >= nrows) r = nrows - 1;
  const float* p = src + r * nf;
#pragma unroll
  for (int jc = 0; jc < 4; ++jc) {
    const int c0 = half * 32 + jc * 8;
    v8h h;
    if (c0 + 8 <= nf) {
      h = cvt8v(*(const float4*)(p + c0), *(const float4*)(p + c0 + 4));
    } else {
#pragma unroll
      for (int q = 0; q < 8; ++q) h[q] = (_Float16)0.0f;
    }
    *(v8h*)(act + row * ACT_LD + c0) = h;
  }
  lds_fence();
}

// ---- register-held gather (overlap global latency with compute) ------------
struct G8 { float4 f[8]; };   // 32 floats: lane's half-row of y[node][64]

__device__ __forceinline__ G8 gather_ld(const float* __restrict__ y,
                                        const int* __restrict__ edges,
                                        long ebase, long nE, int sel, int lane) {
  const int row  = lane >> 1;
  const int half = lane & 1;
  long e = ebase + row;
  if (e >= nE) e = nE - 1;
  const int node = edges[2 * e + sel];
  const float* p = y + (long)node * 64 + half * 32;
  G8 g;
#pragma unroll
  for (int j = 0; j < 8; ++j) g.f[j] = *(const float4*)(p + j * 4);
  return g;
}

__device__ __forceinline__ void gather_st(_Float16* act, const G8& g, int lane) {
  const int row  = lane >> 1;
  const int half = lane & 1;
#pragma unroll
  for (int jc = 0; jc < 4; ++jc) {
    *(v8h*)(act + row * ACT_LD + half * 32 + jc * 8) = cvt8v(g.f[2 * jc], g.f[2 * jc + 1]);
  }
  lds_fence();
}

// 3-layer MLP: DINP->64 (lrelu) -> 64 (lrelu) -> DOUT (linear).
template<int DINP, int DOUT>
__device__ __forceinline__ void mlp3(_Float16* act, const MlpWS& w, int lane, v8f* out) {
  v8f h[4];
  layer_frags<DINP, 64>(act, w.W0, w.b0, lane, h);
  frags_to_act(act, lane, h, 4, true);
  layer_frags<64, 64>(act, w.W1, w.b1, lane, h);
  frags_to_act(act, lane, h, 4, true);
  layer_frags<64, DOUT>(act, w.W2, w.b2, lane, out);
}

// ---- kernels ---------------------------------------------------------------

// y = mlp_f(node_feats)
__global__ void __launch_bounds__(256) k_mlp_f(const float* __restrict__ nfeat,
                                               MlpW w, float* __restrict__ y) {
  extern __shared__ _Float16 smem[];
  _Float16* w0 = smem;                // 32x64 -> 2048 halves
  _Float16* w1 = w0 + 32 * 64;        // 64x64 -> 4096
  _Float16* w2 = w1 + 64 * 64;        // 64x64 -> 4096
  float* bb = (float*)(w2 + 64 * 64); // 3 x 64 f32
  _Float16* actb = (_Float16*)(bb + 192);
  stage_weights(w.W0, w0, 32, 32, 64);
  stage_weights(w.W1, w1, 64, 64, 64);
  stage_weights(w.W2, w2, 64, 64, 64);
  stage_bias(w.b0, bb, 64); stage_bias(w.b1, bb + 64, 64); stage_bias(w.b2, bb + 128, 64);
  __syncthreads();
  const int wid = threadIdx.x >> 5, lane = threadIdx.x & 31;
  _Float16* act = actb + wid * ACT_TILE;
  MlpWS ws{w0, w1, w2, bb, bb + 64, bb + 128};
  for (long tg = blockIdx.x; tg * ROWS_PB < GN_NODES; tg += gridDim.x) {
    long base = tg * ROWS_PB + (long)wid * 16;
    if (base >= GN_NODES) continue;
    stage_rows(nfeat, base, GN_NODES, GN_NFEAT, act, lane);
    v8f o[4];
    mlp3<32, 64>(act, ws, lane, o);
    frags_to_global(y, base, GN_NODES, 64, lane, o, 4);
  }
}

// agg[src] += y[dst] over edges (segment_sum), f32 atomics.
__global__ void __launch_bounds__(256) k_scatter(const float* __restrict__ y,
                                                 const int* __restrict__ edges,
                                                 float* __restrict__ agg) {
  long t = (long)blockIdx.x * blockDim.x + threadIdx.x;
  long e = t >> 6;
  int  c = (int)(t & 63);
  if (e < GN_EDGES) {
    int s = edges[2 * e];
    int d = edges[2 * e + 1];
    atomicAdd(&agg[(long)s * 64 + c], y[(long)d * 64 + c]);
  }
}

// y = mlp_g(agg) + y   (in place; each tile owns its rows)
__global__ void __launch_bounds__(256) k_mlp_g(const float* __restrict__ agg,
                                               float* __restrict__ y, MlpW w) {
  extern __shared__ _Float16 smem[];
  _Float16* w0 = smem;
  _Float16* w1 = w0 + 64 * 64;
  _Float16* w2 = w1 + 64 * 64;
  float* bb = (float*)(w2 + 64 * 64);
  _Float16* actb = (_Float16*)(bb + 192);
  stage_weights(w.W0, w0, 64, 64, 64);
  stage_weights(w.W1, w1, 64, 64, 64);
  stage_weights(w.W2, w2, 64, 64, 64);
  stage_bias(w.b0, bb, 64); stage_bias(w.b1, bb + 64, 64); stage_bias(w.b2, bb + 128, 64);
  __syncthreads();
  const int wid = threadIdx.x >> 5, lane = threadIdx.x & 31;
  _Float16* act = actb + wid * ACT_TILE;
  MlpWS ws{w0, w1, w2, bb, bb + 64, bb + 128};
  for (long tg = blockIdx.x; tg * ROWS_PB < GN_NODES; tg += gridDim.x) {
    long base = tg * ROWS_PB + (long)wid * 16;
    if (base >= GN_NODES) continue;
    stage_rows(agg, base, GN_NODES, 64, act, lane);
    v8f o[4];
    mlp3<64, 64>(act, ws, lane, o);
    const int col0 = lane & 15, row0 = (lane >> 4) << 3;
    for (int nt = 0; nt < 4; ++nt) {
#pragma unroll
      for (int r = 0; r < 8; ++r) {
        long row = base + row0 + r;
        if (row < GN_NODES) {
          long idx = row * 64 + nt * 16 + col0;
          y[idx] = o[nt][r] + y[idx];   // residual
        }
      }
    }
  }
}

// node_outputs = mlp_h(y)
__global__ void __launch_bounds__(256) k_mlp_h(const float* __restrict__ y,
                                               MlpW w, float* __restrict__ out) {
  extern __shared__ _Float16 smem[];
  _Float16* w0 = smem;
  _Float16* w1 = w0 + 64 * 64;
  _Float16* w2 = w1 + 64 * 64;       // 64x32 -> 2048
  float* bb = (float*)(w2 + 64 * 32);
  _Float16* actb = (_Float16*)(bb + 192);
  stage_weights(w.W0, w0, 64, 64, 64);
  stage_weights(w.W1, w1, 64, 64, 64);
  stage_weights(w.W2, w2, 64, 64, 32);
  stage_bias(w.b0, bb, 64); stage_bias(w.b1, bb + 64, 64); stage_bias(w.b2, bb + 128, 32);
  __syncthreads();
  const int wid = threadIdx.x >> 5, lane = threadIdx.x & 31;
  _Float16* act = actb + wid * ACT_TILE;
  MlpWS ws{w0, w1, w2, bb, bb + 64, bb + 128};
  for (long tg = blockIdx.x; tg * ROWS_PB < GN_NODES; tg += gridDim.x) {
    long base = tg * ROWS_PB + (long)wid * 16;
    if (base >= GN_NODES) continue;
    stage_rows(y, base, GN_NODES, 64, act, lane);
    v8f o[4];
    mlp3<64, 32>(act, ws, lane, o);
    frags_to_global(out, base, GN_NODES, GN_NOUT, lane, o, 2);
  }
}

// edge_outputs = mlp_q( mlp_k1(y[src]) + mlp_k2(y[dst]) + mlp_l(edge_feats) )
__global__ void __launch_bounds__(256) k_edge(const float* __restrict__ y,
                                              const int* __restrict__ edges,
                                              const float* __restrict__ ef,
                                              MlpW wk1, MlpW wk2, MlpW wl, MlpW wq,
                                              float* __restrict__ out) {
  extern __shared__ _Float16 smem[];
  _Float16* p = smem;
  _Float16* k1w0 = p; p += 64 * 64;  _Float16* k1w1 = p; p += 64 * 64;  _Float16* k1w2 = p; p += 64 * 64;
  _Float16* k2w0 = p; p += 64 * 64;  _Float16* k2w1 = p; p += 64 * 64;  _Float16* k2w2 = p; p += 64 * 64;
  _Float16* lw0  = p; p += 32 * 64;  _Float16* lw1  = p; p += 64 * 64;  _Float16* lw2  = p; p += 64 * 64;
  _Float16* qw0  = p; p += 64 * 64;  _Float16* qw1  = p; p += 64 * 64;  _Float16* qw2  = p; p += 64 * 16;
  float* bb = (float*)p;             // 12 x 64 f32 slots
  _Float16* actb = (_Float16*)(bb + 12 * 64);
  stage_weights(wk1.W0, k1w0, 64, 64, 64);
  stage_weights(wk1.W1, k1w1, 64, 64, 64);
  stage_weights(wk1.W2, k1w2, 64, 64, 64);
  stage_weights(wk2.W0, k2w0, 64, 64, 64);
  stage_weights(wk2.W1, k2w1, 64, 64, 64);
  stage_weights(wk2.W2, k2w2, 64, 64, 64);
  stage_weights(wl.W0,  lw0,  16, 32, 64);   // pad K 16 -> 32 with zeros
  stage_weights(wl.W1,  lw1,  64, 64, 64);
  stage_weights(wl.W2,  lw2,  64, 64, 64);
  stage_weights(wq.W0,  qw0,  64, 64, 64);
  stage_weights(wq.W1,  qw1,  64, 64, 64);
  stage_weights(wq.W2,  qw2,  64, 64, 16);
  stage_bias(wk1.b0, bb + 0 * 64, 64); stage_bias(wk1.b1, bb + 1 * 64, 64); stage_bias(wk1.b2, bb + 2 * 64, 64);
  stage_bias(wk2.b0, bb + 3 * 64, 64); stage_bias(wk2.b1, bb + 4 * 64, 64); stage_bias(wk2.b2, bb + 5 * 64, 64);
  stage_bias(wl.b0,  bb + 6 * 64, 64); stage_bias(wl.b1,  bb + 7 * 64, 64); stage_bias(wl.b2,  bb + 8 * 64, 64);
  stage_bias(wq.b0,  bb + 9 * 64, 64); stage_bias(wq.b1,  bb + 10 * 64, 64); stage_bias(wq.b2, bb + 11 * 64, 16);
  __syncthreads();
  const int wid = threadIdx.x >> 5, lane = threadIdx.x & 31;
  _Float16* act = actb + wid * ACT_TILE;
  MlpWS ws1{k1w0, k1w1, k1w2, bb + 0 * 64, bb + 1 * 64, bb + 2 * 64};
  MlpWS ws2{k2w0, k2w1, k2w2, bb + 3 * 64, bb + 4 * 64, bb + 5 * 64};
  MlpWS wsl{lw0,  lw1,  lw2,  bb + 6 * 64, bb + 7 * 64, bb + 8 * 64};
  for (long tg = blockIdx.x; tg * ROWS_PB < GN_EDGES; tg += gridDim.x) {
    long base = tg * ROWS_PB + (long)wid * 16;
    if (base >= GN_EDGES) continue;
    v8f sum[4], t[4];
    // Issue both gathers up front: dst data stays in registers across k1's MLP.
    G8 gs = gather_ld(y, edges, base, GN_EDGES, 0, lane);
    G8 gd = gather_ld(y, edges, base, GN_EDGES, 1, lane);
    // Prefetch this tile's edge-feature rows (one 64B line per row).
    {
      long e = base + (lane >> 1);
      if (e >= GN_EDGES) e = GN_EDGES - 1;
      __builtin_prefetch(ef + e * GN_EFEAT, 0, 1);
    }
    // left = k1(y[src])
    gather_st(act, gs, lane);
    mlp3<64, 64>(act, ws1, lane, sum);
    // right = k2(y[dst])  (data already resident in registers)
    gather_st(act, gd, lane);
    mlp3<64, 64>(act, ws2, lane, t);
#pragma unroll
    for (int i = 0; i < 4; ++i) sum[i] += t[i];
    // + l(edge_feats)
    stage_rows(ef, base, GN_EDGES, GN_EFEAT, act, lane);
    mlp3<32, 64>(act, wsl, lane, t);
#pragma unroll
    for (int i = 0; i < 4; ++i) sum[i] += t[i];
    // edge_emb -> q MLP
    frags_to_act(act, lane, sum, 4, false);
    v8f h[4], o[4];
    layer_frags<64, 64>(act, qw0, bb + 9 * 64, lane, h);
    frags_to_act(act, lane, h, 4, true);
    layer_frags<64, 64>(act, qw1, bb + 10 * 64, lane, h);
    frags_to_act(act, lane, h, 4, true);
    layer_frags<64, 16>(act, qw2, bb + 11 * 64, lane, o);
    frags_to_global(out, base, GN_EDGES, GN_EOUT, lane, o, 1);
  }
}

// ---- host ------------------------------------------------------------------
extern "C" void kernel_launch(void* const* d_in, const int* in_sizes, int n_in,
                              void* d_out, int out_size, void* d_ws, size_t ws_size,
                              hipStream_t stream) {
  const float* node_feats = (const float*)d_in[0];
  const int*   edges      = (const int*)d_in[1];
  const float* edge_feats = (const float*)d_in[2];
  auto M = [&](int i) {
    return MlpW{(const float*)d_in[i],     (const float*)d_in[i + 1],
                (const float*)d_in[i + 2], (const float*)d_in[i + 3],
                (const float*)d_in[i + 4], (const float*)d_in[i + 5]};
  };
  // params flatten order (alphabetical == insertion): f,g,h,k1,k2,l,q; each W0,b0,W1,b1,W2,b2
  MlpW wf = M(3),  wg = M(9),  wh = M(15);
  MlpW wk1 = M(21), wk2 = M(27), wl = M(33), wq = M(39);

  float* y   = (float*)d_ws;                       // [50000][64]
  float* agg = y + (size_t)GN_NODES * 64;          // [50000][64]
  float* node_out = (float*)d_out;                 // [50000][32]
  float* edge_out = node_out + (size_t)GN_NODES * GN_NOUT;  // [800000][16]

  const dim3 blk(256);
  const size_t actB  = (size_t)WAVES_PB * ACT_TILE * 2;   // bytes (8 wave tiles)
  const size_t shm_f = (size_t)(32*64 + 2*64*64) * 2 + 192 * 4 + actB;
  const size_t shm_g = (size_t)(3*64*64) * 2 + 192 * 4 + actB;
  const size_t shm_h = (size_t)(2*64*64 + 64*32) * 2 + 192 * 4 + actB;
  const size_t shm_e = (size_t)(6*64*64 + 32*64 + 2*64*64 + 2*64*64 + 64*16) * 2
                       + 12 * 64 * 4 + actB;

  const int nodeBlocks = (GN_NODES + ROWS_PB - 1) / ROWS_PB;   // 391
  k_mlp_f<<<nodeBlocks, blk, shm_f, stream>>>(node_feats, wf, y);
  for (int s = 0; s < 2; ++s) {
    hipMemsetAsync(agg, 0, (size_t)GN_NODES * 64 * sizeof(float), stream);
    long threads = (long)GN_EDGES * 64;
    k_scatter<<<(int)((threads + 255) / 256), blk, 0, stream>>>(y, edges, agg);
    k_mlp_g<<<nodeBlocks, blk, shm_g, stream>>>(agg, y, wg);
  }
  k_mlp_h<<<nodeBlocks, blk, shm_h, stream>>>(y, wh, node_out);
  k_edge<<<1250, blk, shm_e, stream>>>(y, edges, edge_feats, wk1, wk2, wl, wq, edge_out);
}
